// LSTM_609885356117
// MI455X (gfx1250) — matrix operand order repaired and checked by reading, not compiled, not run on hardware
//
#include <hip/hip_runtime.h>
#include <hip/hip_bf16.h>
#include <math.h>

// ============================================================================
// LSTM forward for MI455X (gfx1250, wave32, WMMA).
//
// Phase 1 (parallel): Gx[t] = x_t @ Wx  for all 512 steps at once
//   (bf16 WMMA, f32 out, 268 MB scratch) -- no recurrent dependence.
// Phase 2 (serial, 512 launches): per step only the recurrent half
//   [32 x 1024] @ [1024 x 4096] in bf16 WMMA; Wh (8 MB bf16) L2-resident.
// Fallback (small ws): full-K [32 x 2048] step kernel (no Gx scratch).
//
// ws layout (bytes):
//   Wp : 16 MB  bf16 weights, WMMA-B fragment layout [256 nt][64 kt][32 ln][16]
//   Xp : 32 MB  bf16 x, WMMA-A fragment layout [512 t][2 mt][32 kt][32 ln][16]
//   Hp : 128 KB double-buffered bf16 h in A fragment layout
//   Gx : 268 MB f32 precomputed x-projection [512 t][32 b][4096]  (optional)
//
// d_out (float): h_last[32*1024] | h_states[512*32*1024] | c_states[512*32*1024]
// ============================================================================

typedef __attribute__((ext_vector_type(16))) __bf16 v16bf;
typedef __attribute__((ext_vector_type(8)))  float  v8f;

__device__ __forceinline__ unsigned short f2bf(float f) {
  unsigned u = __float_as_uint(f);
  u += 0x7fffu + ((u >> 16) & 1u);          // round-to-nearest-even
  return (unsigned short)(u >> 16);
}

__device__ __forceinline__ v8f wmma_bf16(v16bf a, v16bf b, v8f c) {
  return __builtin_amdgcn_wmma_f32_16x16x32_bf16(
      /*neg_a=*/false, a, /*neg_b=*/false, b,
      /*c_mod=*/(short)0, c, /*reuse_a=*/false, /*reuse_b=*/false);
}

// ---------------------------------------------------------------------------
// Pack W = concat([Wf,Wi,Wo,Wc], axis=1) into bf16 WMMA-B fragment layout:
//   B tile 32x16: column N = lane&15, K = (e&7) + (lane>>4)*8 + (e>>3)*16.
// ---------------------------------------------------------------------------
__global__ void conv_w_kernel(const float* __restrict__ Wf, const float* __restrict__ Wi,
                              const float* __restrict__ Wo, const float* __restrict__ Wc,
                              unsigned short* __restrict__ Wp) {
  unsigned gid   = blockIdx.x * 256u + threadIdx.x;     // 8,388,608 total
  unsigned e     = gid & 15u;
  unsigned lane  = (gid >> 4) & 31u;
  unsigned ktile = (gid >> 9) & 63u;
  unsigned ntile = gid >> 15;                           // 0..255
  unsigned K     = (e & 7u) + ((lane >> 4) << 3) + ((e >> 3) << 4);
  unsigned N     = ntile * 16u + (lane & 15u);          // 0..4095 across 4 gates
  unsigned gk    = ktile * 32u + K;                     // 0..2047 (row of W)
  unsigned gate  = N >> 10;
  unsigned col   = N & 1023u;
  const float* Wg = (gate == 0) ? Wf : (gate == 1) ? Wi : (gate == 2) ? Wo : Wc;
  Wp[gid] = f2bf(Wg[(size_t)gk * 1024u + col]);
}

// ---------------------------------------------------------------------------
// Pack x [32,512,1024] into bf16 WMMA-A fragment layout per timestep.
//   A tile 16x32 (ISA 7.12.2): kk = (e&7) + half*8 + ((e>>3)<<4).
// ---------------------------------------------------------------------------
__global__ void conv_x_kernel(const float* __restrict__ x,
                              unsigned short* __restrict__ Xp) {
  unsigned gid   = blockIdx.x * 256u + threadIdx.x;     // 16,777,216 total
  unsigned e     = gid & 15u;
  unsigned lane  = (gid >> 4) & 31u;
  unsigned ktile = (gid >> 9) & 31u;
  unsigned mtile = (gid >> 14) & 1u;
  unsigned t     = gid >> 15;                           // 0..511
  unsigned half  = lane >> 4;
  unsigned kk    = (e & 7u) + (half << 3) + ((e >> 3) << 4);
  unsigned m     = mtile * 16u + (lane & 15u);          // batch 0..31
  unsigned kp    = ktile * 32u + kk;                    // 0..1023 (input feature)
  Xp[gid] = f2bf(x[((size_t)m * 512u + t) * 1024u + kp]);
}

// Zero both h fragment buffers (h0 = 0).
__global__ void init_h_kernel(unsigned int* __restrict__ Hp32) {
  unsigned i = blockIdx.x * 256u + threadIdx.x;         // 16384 threads
  Hp32[i] = 0u;
  Hp32[i + 16384u] = 0u;
}

// ---------------------------------------------------------------------------
// Phase 1: Gx[t][b][n] = sum_{k<1024} x[b][t][k] * W[k][n]  for all t.
// Grid (512, 32) x 256 threads; wave w handles n-tile blockIdx.y*8 + w,
// both M-tiles, full 32 x-k-tiles, depth-2 pipelined.
// ---------------------------------------------------------------------------
__global__ __launch_bounds__(256) void xproj_kernel(
    const unsigned short* __restrict__ Wp,
    const unsigned short* __restrict__ Xp,
    float* __restrict__ Gx)
{
  const int lane  = threadIdx.x & 31;
  const int wave  = threadIdx.x >> 5;              // 0..7
  const int t     = blockIdx.x;                    // 0..511
  const int ntile = blockIdx.y * 8 + wave;         // 0..255

  const unsigned short* Ab = Xp + (size_t)t * 32768u;
  const unsigned short* Bb = Wp + (size_t)ntile * 64u * 512u;  // x k-tiles 0..31

  auto ldA = [&](int mt, int i) -> v16bf {
    return *(const v16bf*)(Ab + (((size_t)mt * 32 + i) * 32 + lane) * 16);
  };
  auto ldB = [&](int i) -> v16bf {
    return *(const v16bf*)(Bb + ((size_t)i * 32 + lane) * 16);
  };

  v8f acc0 = {}; v8f acc1 = {};
  v16bf a0A = ldA(0, 0), a1A = ldA(1, 0), bA = ldB(0);
  v16bf a0B = ldA(0, 1), a1B = ldA(1, 1), bB = ldB(1);

  #pragma unroll 4
  for (int kt = 0; kt < 32; kt += 2) {
    const int kp = (kt + 6 < 32) ? kt + 6 : 31;
    __builtin_prefetch((const void*)(Bb + ((size_t)kp * 32 + lane) * 16), 0, 3);
    const int k2 = (kt + 2 < 32) ? kt + 2 : kt;
    const int k3 = (kt + 3 < 32) ? kt + 3 : kt + 1;
    v16bf na0A = ldA(0, k2), na1A = ldA(1, k2), nbA = ldB(k2);
    acc0 = wmma_bf16(a0A, bA, acc0);
    acc1 = wmma_bf16(a1A, bA, acc1);
    v16bf na0B = ldA(0, k3), na1B = ldA(1, k3), nbB = ldB(k3);
    acc0 = wmma_bf16(a0B, bB, acc0);
    acc1 = wmma_bf16(a1B, bB, acc1);
    a0A = na0A; a1A = na1A; bA = nbA;
    a0B = na0B; a1B = na1B; bB = nbB;
  }

  // C layout: VGPR r -> row r (lanes 0-15) / r+8 (lanes 16-31), col = lane&15.
  const int col = lane & 15;
  const int rb  = (lane >> 4) << 3;
  #pragma unroll
  for (int r = 0; r < 8; ++r) {
    const int b0 = r + rb;
    Gx[((size_t)t * 32 + b0)      * 4096u + (size_t)ntile * 16 + col] = acc0[r];
    Gx[((size_t)t * 32 + b0 + 16) * 4096u + (size_t)ntile * 16 + col] = acc1[r];
  }
}

// ---------------------------------------------------------------------------
// Phase 2: one LSTM timestep. 64 blocks x 256 threads (8 waves):
// wave = (gate, k-half). XPROJ=true: K = recurrent 1024 only (16 k-tiles per
// wave), epilogue adds precomputed Gx. XPROJ=false: full K = 2048 (32 k-tiles
// per wave). Partials summed in the LDS gate exchange.
// ---------------------------------------------------------------------------
template <bool XPROJ>
__global__ __launch_bounds__(256) void lstm_step_kernel(
    const unsigned short* __restrict__ Wp,
    const unsigned short* __restrict__ Xp,
    unsigned short*       __restrict__ Hp,
    const float* __restrict__ Gx,
    const float* __restrict__ bfv, const float* __restrict__ biv,
    const float* __restrict__ bov, const float* __restrict__ bcv,
    float* __restrict__ out, int t)
{
  const int lane = threadIdx.x & 31;
  const int wave = threadIdx.x >> 5;        // 0..7
  const int gate = wave & 3;                // 0..3  (f, i, o, g)
  const int kh   = wave >> 2;               // k-half within this kernel's K
  const int wg   = blockIdx.x;              // 0..63 -> hidden cols [wg*16, +16)
  const int ntg  = gate * 64 + wg;          // global 16-col tile in [0,256)

  const unsigned short* HpCur = Hp + (size_t)( t      & 1) * 32768u;
  unsigned short*       HpNxt = Hp + (size_t)((t + 1) & 1) * 32768u;
  const unsigned short* xb    = Xp + (size_t)t * 32768u;

  constexpr int NKT = XPROJ ? 16 : 32;      // k-tiles per wave
  const unsigned short* Ab;                 // A fragments [mt][32 kt][lane][16]
  int at0, bt0;                             // A-local / B-global k-tile start
  if (XPROJ) { Ab = HpCur;           at0 = kh * 16; bt0 = 32 + kh * 16; }
  else       { Ab = kh ? HpCur : xb; at0 = 0;       bt0 = kh * 32;      }
  const unsigned short* Bb = Wp + ((size_t)ntg * 64 + bt0) * 512u;

  auto ldA = [&](int mt, int i) -> v16bf {
    return *(const v16bf*)(Ab + (((size_t)mt * 32 + at0 + i) * 32 + lane) * 16);
  };
  auto ldB = [&](int i) -> v16bf {
    return *(const v16bf*)(Bb + ((size_t)i * 32 + lane) * 16);
  };

  v8f acc0 = {}; v8f acc1 = {};
  v16bf a0A = ldA(0, 0), a1A = ldA(1, 0), bA = ldB(0);
  v16bf a0B = ldA(0, 1), a1B = ldA(1, 1), bB = ldB(1);

  #pragma unroll 4
  for (int kt = 0; kt < NKT; kt += 2) {
    const int kp = (kt + 6 < NKT) ? kt + 6 : NKT - 1;   // L1 warm for B stream
    __builtin_prefetch((const void*)(Bb + ((size_t)kp * 32 + lane) * 16), 0, 3);
    const int k2 = (kt + 2 < NKT) ? kt + 2 : kt;        // tail prefetches are
    const int k3 = (kt + 3 < NKT) ? kt + 3 : kt + 1;    // redundant but safe
    v16bf na0A = ldA(0, k2), na1A = ldA(1, k2), nbA = ldB(k2);
    acc0 = wmma_bf16(a0A, bA, acc0);
    acc1 = wmma_bf16(a1A, bA, acc1);
    v16bf na0B = ldA(0, k3), na1B = ldA(1, k3), nbB = ldB(k3);
    acc0 = wmma_bf16(a0B, bB, acc0);
    acc1 = wmma_bf16(a1B, bB, acc1);
    a0A = na0A; a1A = na1A; bA = nbA;
    a0B = na0B; a1B = na1B; bB = nbB;
  }

  // ---- cross-gate exchange + K-half reduction (C layout per ISA 7.12.2).
  __shared__ float lds[8][32][16];
  {
    const int col = lane & 15;
    const int rb  = (lane >> 4) << 3;
    #pragma unroll
    for (int r = 0; r < 8; ++r) {
      lds[wave][r + rb][col]      = acc0[r];
      lds[wave][r + rb + 16][col] = acc1[r];
    }
  }
  __syncthreads();

  const size_t H_OFF = 32768u;
  const size_t C_OFF = 32768u + (size_t)512 * 32768u;

  #pragma unroll
  for (int j = 0; j < 2; ++j) {
    const int idx = threadIdx.x + j * 256;   // 0..511 = 32 batches x 16 cols
    const int b   = idx >> 4;
    const int col = idx & 15;
    const int hc  = wg * 16 + col;

    float fv = lds[0][b][col] + lds[4][b][col] + bfv[hc];
    float iv = lds[1][b][col] + lds[5][b][col] + biv[hc];
    float ov = lds[2][b][col] + lds[6][b][col] + bov[hc];
    float gv = lds[3][b][col] + lds[7][b][col] + bcv[hc];
    if (XPROJ) {
      const float* gx = Gx + ((size_t)t * 32 + b) * 4096u + hc;
      fv += gx[0];
      iv += gx[1024];
      ov += gx[2048];
      gv += gx[3072];
    }
    fv = 1.0f / (1.0f + __expf(-fv));
    iv = 1.0f / (1.0f + __expf(-iv));
    ov = 1.0f / (1.0f + __expf(-ov));
    gv = tanhf(gv);

    const size_t bh = (size_t)b * 1024u + hc;
    float cp = (t == 0) ? 0.0f : out[C_OFF + (size_t)(t - 1) * 32768u + bh];
    float c  = fv * cp + iv * gv;
    float h  = ov * tanhf(c);

    out[C_OFF + (size_t)t * 32768u + bh] = c;
    out[H_OFF + (size_t)t * 32768u + bh] = h;
    if (t == 511) out[bh] = h;

    // h feedback in WMMA-A fragment layout for the next step (k' = hc).
    const int mt   = b >> 4;
    const int lm   = b & 15;
    const int ktl  = hc >> 5;
    const int kk   = hc & 31;
    const int half = (kk >> 3) & 1;
    const int e    = (kk & 7) + ((kk >> 4) << 3);
    const int ln   = lm + half * 16;
    HpNxt[(((size_t)mt * 32 + ktl) * 32 + ln) * 16 + e] = f2bf(h);
  }
}

// ============================================================================
extern "C" void kernel_launch(void* const* d_in, const int* in_sizes, int n_in,
                              void* d_out, int out_size, void* d_ws, size_t ws_size,
                              hipStream_t stream) {
  const float* x  = (const float*)d_in[0];
  const float* Wf = (const float*)d_in[1];
  const float* bf = (const float*)d_in[2];
  const float* Wi = (const float*)d_in[3];
  const float* bi = (const float*)d_in[4];
  const float* Wo = (const float*)d_in[5];
  const float* bo = (const float*)d_in[6];
  const float* Wc = (const float*)d_in[7];
  const float* bc = (const float*)d_in[8];

  unsigned short* Wp = (unsigned short*)d_ws;
  unsigned short* Xp = Wp + 8388608u;               // +16 MB
  unsigned short* Hp = Xp + 16777216u;              // +32 MB
  const size_t BASE_BYTES = 50462720u;              // 16M + 32M + 128K
  float* Gx = (float*)((char*)d_ws + BASE_BYTES);   // 268,435,456 bytes
  const bool use_xproj = ws_size >= BASE_BYTES + (size_t)268435456u;
  float* out = (float*)d_out;

  conv_w_kernel<<<32768, 256, 0, stream>>>(Wf, Wi, Wo, Wc, Wp);
  conv_x_kernel<<<65536, 256, 0, stream>>>(x, Xp);
  init_h_kernel<<<64, 256, 0, stream>>>((unsigned int*)Hp);

  if (use_xproj) {
    xproj_kernel<<<dim3(512, 32), 256, 0, stream>>>(Wp, Xp, Gx);
    for (int t = 0; t < 512; ++t) {
      lstm_step_kernel<true><<<64, 256, 0, stream>>>(
          Wp, Xp, Hp, Gx, bf, bi, bo, bc, out, t);
    }
  } else {
    for (int t = 0; t < 512; ++t) {
      lstm_step_kernel<false><<<64, 256, 0, stream>>>(
          Wp, Xp, Hp, nullptr, bf, bi, bo, bc, out, t);
    }
  }
}
